// ShuffleNetUnitA_52587579572443
// MI455X (gfx1250) — matrix operand order repaired and checked by reading, not compiled, not run on hardware
//
#include <hip/hip_runtime.h>
#include <hip/hip_bf16.h>

typedef __attribute__((ext_vector_type(16))) _Float16 v16h;
typedef __attribute__((ext_vector_type(8)))  _Float16 v8h;
typedef __attribute__((ext_vector_type(8)))  float    v8f;
typedef __attribute__((ext_vector_type(4)))  float    v4f;

#define EPS 1e-3f
#define TIN 14           // inner tile (14x14); halo tile is 16x16 = 256 px = 16 wave-tiles
#define GRP 8

// LDS carve-up (bytes)
#define OFF_H1   0                        // 256 px * 128 ch * f16 = 65536
#define OFF_W1   65536                    // 16 frags * 512 f16    = 16384
#define OFF_W2   (65536 + 16384)          // 32 frags * 512 f16    = 32768
#define OFF_BN   (65536 + 16384 + 32768)  // S1,T1,S23,T23 (128 ea) + S4,T4 (512 ea) = 6144
#define OFF_DW   (OFF_BN + (4*128 + 2*512) * 4)
#define SMEM_BYTES (OFF_DW + 9 * 128 * 4) // + depthwise weights 4608 -> 125440 total

__global__ __launch_bounds__(256)
void shufflenet_unit_fused(
    const float* __restrict__ x,
    const float* __restrict__ w1, const float* __restrict__ b1,
    const float* __restrict__ dww, const float* __restrict__ dwb,
    const float* __restrict__ w2, const float* __restrict__ b2,
    const float* __restrict__ bn1g, const float* __restrict__ bn1b,
    const float* __restrict__ bn1m, const float* __restrict__ bn1v,
    const float* __restrict__ bn2g, const float* __restrict__ bn2b,
    const float* __restrict__ bn2m, const float* __restrict__ bn2v,
    const float* __restrict__ bn3g, const float* __restrict__ bn3b,
    const float* __restrict__ bn3m, const float* __restrict__ bn3v,
    const float* __restrict__ bn4g, const float* __restrict__ bn4b,
    const float* __restrict__ bn4m, const float* __restrict__ bn4v,
    float* __restrict__ out)
{
    extern __shared__ unsigned char smem[];
    _Float16* sh1 = (_Float16*)(smem + OFF_H1);   // shuffled conv1 output, halo tile
    _Float16* sw1 = (_Float16*)(smem + OFF_W1);   // conv1 B fragments [g*2+s][lane*16+e]
    _Float16* sw2 = (_Float16*)(smem + OFF_W2);   // conv2 B fragments [g*4+t][lane*16+e]
    float* sS1  = (float*)(smem + OFF_BN);
    float* sT1  = sS1  + 128;
    float* sS23 = sT1  + 128;
    float* sT23 = sS23 + 128;
    float* sS4  = sT23 + 128;
    float* sT4  = sS4  + 512;
    float* sdw  = (float*)(smem + OFF_DW);        // depthwise weights [tap][ch]

    const int tid  = threadIdx.x;
    const int lane = tid & 31;
    const int wv   = tid >> 5;
    const int bx = blockIdx.x, by = blockIdx.y, img = blockIdx.z;

    // ---------- Phase 0: fold BN chains, swizzle weights into WMMA B layout ----------
    for (int c = tid; c < 128; c += 256) {
        float s1 = bn1g[c] * rsqrtf(bn1v[c] + EPS);
        sS1[c] = s1;
        sT1[c] = bn1b[c] - bn1m[c] * s1 + b1[c] * s1;
        float s2 = bn2g[c] * rsqrtf(bn2v[c] + EPS);
        float t2 = bn2b[c] - bn2m[c] * s2;
        float s3 = bn3g[c] * rsqrtf(bn3v[c] + EPS);
        float t3 = bn3b[c] - bn3m[c] * s3;
        sS23[c] = s2 * s3;
        sT23[c] = (dwb[c] * s2 + t2) * s3 + t3;
    }
    for (int c = tid; c < 512; c += 256) {
        float s4 = bn4g[c] * rsqrtf(bn4v[c] + EPS);
        sS4[c] = s4;
        sT4[c] = bn4b[c] - bn4m[c] * s4 + b2[c] * s4;
    }
    for (int i = tid; i < 9 * 128; i += 256) sdw[i] = dww[i];

    // w1 [8][64][16] -> B frag (32x16 f16): lane n holds K=(e<8?0:16)+(n<16?0:8)+(e&7), col n&15
    for (int i = tid; i < 16 * 512; i += 256) {
        int frag = i >> 9, idx = i & 511;
        int ln = idx >> 4, e = idx & 15;
        int g = frag >> 1, s = frag & 1;
        int c = s * 32 + ((e & 8) ? 16 : 0) + ((ln & 16) ? 8 : 0) + (e & 7);
        sw1[i] = (_Float16)w1[(g * 64 + c) * 16 + (ln & 15)];
    }
    // w2 [8][16][64] -> B frag with K=16..31 zero-padded
    for (int i = tid; i < 32 * 512; i += 256) {
        int frag = i >> 9, idx = i & 511;
        int ln = idx >> 4, e = idx & 15;
        int g = frag >> 2, t = frag & 3;
        float val = 0.f;
        if (!(e & 8)) val = w2[(g * 16 + ((ln & 16) ? 8 : 0) + e) * 64 + t * 16 + (ln & 15)];
        sw2[i] = (_Float16)val;
    }
    __syncthreads();

    // ---------- Phase 1: group conv1x1 #1 + BN1 + ReLU + channel shuffle -> LDS f16 ----------
    const int y0 = by * TIN - 1;   // halo origin
    const int x0 = bx * TIN - 1;
    for (int tile = wv; tile < 16; tile += 8) {
        // this lane's A-row pixel (halo coords)
        int p  = tile * 16 + (lane & 15);
        int gy = y0 + (p >> 4), gx = x0 + (p & 15);
        int cgy = gy < 0 ? 0 : (gy > 55 ? 55 : gy);
        int cgx = gx < 0 ? 0 : (gx > 55 ? 55 : gx);
        const float* xp = x + (((long)img * 56 + cgy) * 56 + cgx) * 512;
        const int koff  = (lane & 16) ? 8 : 0;

        // D-matrix row pixels + validity (SAME padding => OOB pixels become 0)
        const int prow0 = tile * 16 + ((lane & 16) ? 8 : 0);
        unsigned vmask = 0u;
        #pragma unroll
        for (int v = 0; v < 8; ++v) {
            int pp = prow0 + v;
            int py = y0 + (pp >> 4), px = x0 + (pp & 15);
            if (py >= 0 && py < 56 && px >= 0 && px < 56) vmask |= (1u << v);
        }

        for (int g = 0; g < GRP; ++g) {
            v8f acc = {};
            #pragma unroll
            for (int s = 0; s < 2; ++s) {
                const int clo = g * 64 + s * 32 + koff;
                v4f f0 = *(const v4f*)(xp + clo);
                v4f f1 = *(const v4f*)(xp + clo + 4);
                v4f f2 = *(const v4f*)(xp + clo + 16);
                v4f f3 = *(const v4f*)(xp + clo + 20);
                v16h a;
                #pragma unroll
                for (int e = 0; e < 4; ++e) {
                    a[e]      = (_Float16)f0[e];
                    a[e + 4]  = (_Float16)f1[e];
                    a[e + 8]  = (_Float16)f2[e];
                    a[e + 12] = (_Float16)f3[e];
                }
                v16h bm = *(const v16h*)(sw1 + (g * 2 + s) * 512 + lane * 16);
                acc = __builtin_amdgcn_wmma_f32_16x16x32_f16(false, a, false, bm,
                                                             (short)0, acc, false, false);
            }
            const int d = lane & 15;
            const float S = sS1[g * 16 + d];
            const float T = sT1[g * 16 + d];
            const int o = d * 8 + g;            // channel shuffle (g,d) -> d*8+g
            #pragma unroll
            for (int v = 0; v < 8; ++v) {
                float hv = fmaxf(acc[v] * S + T, 0.f);
                if (!((vmask >> v) & 1u)) hv = 0.f;
                sh1[(prow0 + v) * 128 + o] = (_Float16)hv;
            }
        }
    }
    __syncthreads();

    // ---------- Phase 2: depthwise3x3 + BN2*BN3 (into A frag) + group conv1x1 #2
    //            + BN4 + residual + ReLU -> global ----------
    for (int tile = wv; tile < 13; tile += 8) {       // ceil(196/16)
        int ip = tile * 16 + (lane & 15);
        if (ip > 195) ip = 195;                       // clamp: rows masked at store
        const int hy = ip / 14 + 1, hx = ip % 14 + 1; // halo coords
        const int koff = (lane & 16) ? 8 : 0;

        const int prow0 = tile * 16 + ((lane & 16) ? 8 : 0);
        long baseaddr[8];
        unsigned pmask = 0u;
        #pragma unroll
        for (int v = 0; v < 8; ++v) {
            int pp = prow0 + v;
            if (pp < 196) {
                pmask |= (1u << v);
                int py = by * TIN + pp / 14, px = bx * TIN + pp % 14;
                baseaddr[v] = (((long)img * 56 + py) * 56 + px) * 512;
            } else baseaddr[v] = 0;
        }

        for (int g = 0; g < GRP; ++g) {
            const int c0 = g * 16 + koff;
            float hd[8] = {0, 0, 0, 0, 0, 0, 0, 0};
            #pragma unroll
            for (int t9 = 0; t9 < 9; ++t9) {
                const int ky = t9 / 3, kx = t9 % 3;
                const _Float16* src = sh1 + ((hy + ky - 1) * 16 + (hx + kx - 1)) * 128 + c0;
                v8h hv = *(const v8h*)src;
                const float* wt = sdw + t9 * 128 + c0;
                #pragma unroll
                for (int e = 0; e < 8; ++e) hd[e] += (float)hv[e] * wt[e];
            }
            v16h a;
            #pragma unroll
            for (int e = 0; e < 8; ++e) {
                float z = hd[e] * sS23[c0 + e] + sT23[c0 + e];
                a[e]     = (_Float16)z;
                a[e + 8] = (_Float16)0.f;    // K=16..31 zero-pad
            }
            #pragma unroll
            for (int t = 0; t < 4; ++t) {
                v16h bm = *(const v16h*)(sw2 + (g * 4 + t) * 512 + lane * 16);
                v8f acc = {};
                acc = __builtin_amdgcn_wmma_f32_16x16x32_f16(false, a, false, bm,
                                                             (short)0, acc, false, false);
                const int oc = g * 64 + t * 16 + (lane & 15);
                const float S = sS4[oc];
                const float T = sT4[oc];
                #pragma unroll
                for (int v = 0; v < 8; ++v) {
                    if ((pmask >> v) & 1u) {
                        long xi = baseaddr[v] + oc;
                        out[xi] = fmaxf(x[xi] + acc[v] * S + T, 0.f);
                    }
                }
            }
        }
    }
}

extern "C" void kernel_launch(void* const* d_in, const int* in_sizes, int n_in,
                              void* d_out, int out_size, void* d_ws, size_t ws_size,
                              hipStream_t stream) {
    (void)in_sizes; (void)n_in; (void)out_size; (void)d_ws; (void)ws_size;
    const float* x    = (const float*)d_in[0];
    const float* w1   = (const float*)d_in[1];
    const float* b1   = (const float*)d_in[2];
    const float* dww  = (const float*)d_in[3];
    const float* dwb  = (const float*)d_in[4];
    const float* w2   = (const float*)d_in[5];
    const float* b2   = (const float*)d_in[6];
    const float* bn1g = (const float*)d_in[7];
    const float* bn1b = (const float*)d_in[8];
    const float* bn1m = (const float*)d_in[9];
    const float* bn1v = (const float*)d_in[10];
    const float* bn2g = (const float*)d_in[11];
    const float* bn2b = (const float*)d_in[12];
    const float* bn2m = (const float*)d_in[13];
    const float* bn2v = (const float*)d_in[14];
    const float* bn3g = (const float*)d_in[15];
    const float* bn3b = (const float*)d_in[16];
    const float* bn3m = (const float*)d_in[17];
    const float* bn3v = (const float*)d_in[18];
    const float* bn4g = (const float*)d_in[19];
    const float* bn4b = (const float*)d_in[20];
    const float* bn4m = (const float*)d_in[21];
    const float* bn4v = (const float*)d_in[22];
    float* out = (float*)d_out;

    dim3 grid(4, 4, 32);   // 4x4 spatial tiles of 14x14, 32 images
    dim3 block(256);       // 8 wave32
    shufflenet_unit_fused<<<grid, block, SMEM_BYTES, stream>>>(
        x, w1, b1, dww, dwb, w2, b2,
        bn1g, bn1b, bn1m, bn1v,
        bn2g, bn2b, bn2m, bn2v,
        bn3g, bn3b, bn3m, bn3v,
        bn4g, bn4b, bn4m, bn4v,
        out);
}
